// MoELayer_60790967107614
// MI455X (gfx1250) — compile-verified, matrix operand order checked by reading, not run on hardware
//
#include <hip/hip_runtime.h>

// ---------------------------------------------------------------------------
// MoE layer for MI455X (gfx1250): routed top-2 grouped GEMM on bf16 WMMA.
// N=4096 tokens, D=512, H=2048, E=8 experts, K=2.
// ---------------------------------------------------------------------------

#define MOE_N 4096
#define MOE_D 512
#define MOE_H 2048
#define MOE_E 8
#define MOE_K 2
#define TILE_M 32                    // tokens per GEMM workgroup
#define TILES_PER_E (MOE_N / TILE_M) // worst case: one expert gets every token
#define NWAVE 8                      // 256 threads = 8 wave32

typedef __attribute__((ext_vector_type(16))) __bf16 v16bf;
typedef __attribute__((ext_vector_type(8)))  __bf16 v8bf;
typedef __attribute__((ext_vector_type(8)))  float  v8f;

__device__ __forceinline__ v16bf frag_from(const __bf16* p) {
    // p is 16B-aligned; elements [0..7] and [16..23] form one 16-bit A/B frag
    // (lane-dependent k-half is folded into p by the caller).
    uint4 lo = *(const uint4*)(p);
    uint4 hi = *(const uint4*)(p + 16);
    v8bf a = __builtin_bit_cast(v8bf, lo);
    v8bf b = __builtin_bit_cast(v8bf, hi);
    return __builtin_shufflevector(a, b, 0,1,2,3,4,5,6,7,8,9,10,11,12,13,14,15);
}

__device__ __forceinline__ v8f wmma_bf16(v16bf a, v16bf b, v8f c) {
    return __builtin_amdgcn_wmma_f32_16x16x32_bf16(
        /*neg_a=*/false, a, /*neg_b=*/false, b,
        /*c_mod=*/(short)0, c, /*reuse_a=*/false, /*reuse_b=*/false);
}

// ---------------------------------------------------------------------------
// 1) zero output (combined [N,D] + usage [E]) and expert counts
// ---------------------------------------------------------------------------
__global__ void init_zero_kernel(float* out, int out_elems, int* counts) {
    int i = blockIdx.x * blockDim.x + threadIdx.x;
    if (i < out_elems) out[i] = 0.0f;
    if (blockIdx.x == 0 && threadIdx.x < MOE_E) counts[threadIdx.x] = 0;
}

// ---------------------------------------------------------------------------
// 2) x (f32) -> x (bf16)
// ---------------------------------------------------------------------------
__global__ void cvt_x_kernel(const float* __restrict__ x, __bf16* __restrict__ xb) {
    int i = blockIdx.x * blockDim.x + threadIdx.x;
    if (i < MOE_N * MOE_D) xb[i] = (__bf16)x[i];
}

// ---------------------------------------------------------------------------
// 3) per-expert transpose + f32->bf16: src[e][R][C] -> dst[e][C][R]
//    (gives K-contiguous B operands for WMMA; b128 loads from L2)
// ---------------------------------------------------------------------------
__global__ void transpose_cvt_kernel(const float* __restrict__ src,
                                     __bf16* __restrict__ dst, int R, int C) {
    __shared__ float tile[32][33];
    const int e = blockIdx.z;
    const float* s = src + (size_t)e * R * C;
    __bf16*     d = dst + (size_t)e * R * C;
    const int x  = blockIdx.x * 32 + threadIdx.x;  // col in C
    const int y0 = blockIdx.y * 32;                // row base in R
    #pragma unroll
    for (int j = threadIdx.y; j < 32; j += 8)
        tile[j][threadIdx.x] = s[(size_t)(y0 + j) * C + x];
    __syncthreads();
    const int x2 = y0 + threadIdx.x;               // col in dst row (length R)
    const int y2 = blockIdx.x * 32;                // row base in dst (length C)
    #pragma unroll
    for (int j = threadIdx.y; j < 32; j += 8)
        d[(size_t)(y2 + j) * R + x2] = (__bf16)tile[threadIdx.x][j];
}

// ---------------------------------------------------------------------------
// 4) gating: one wave32 per token. logits = x@Wg + bg; top-2; renormalize.
// ---------------------------------------------------------------------------
__global__ void gate_kernel(const float* __restrict__ x,
                            const float* __restrict__ Wg,
                            const float* __restrict__ bg,
                            int* __restrict__ top_i, float* __restrict__ top_w,
                            int* __restrict__ counts, float* __restrict__ usage) {
    const int wave  = threadIdx.x >> 5;
    const int lane  = threadIdx.x & 31;
    const int token = blockIdx.x * (blockDim.x >> 5) + wave;
    if (token >= MOE_N) return;

    float acc[MOE_E];
    #pragma unroll
    for (int e = 0; e < MOE_E; ++e) acc[e] = 0.0f;

    const float* xr = x + (size_t)token * MOE_D;
    #pragma unroll 4
    for (int i = 0; i < MOE_D / 32; ++i) {
        const int dd = lane + 32 * i;
        const float xv = xr[dd];
        const float4* wg = (const float4*)(Wg + (size_t)dd * MOE_E);
        float4 w0 = wg[0], w1 = wg[1];
        acc[0] += xv * w0.x; acc[1] += xv * w0.y;
        acc[2] += xv * w0.z; acc[3] += xv * w0.w;
        acc[4] += xv * w1.x; acc[5] += xv * w1.y;
        acc[6] += xv * w1.z; acc[7] += xv * w1.w;
    }
    #pragma unroll
    for (int e = 0; e < MOE_E; ++e)
        #pragma unroll
        for (int off = 16; off >= 1; off >>= 1)
            acc[e] += __shfl_down(acc[e], off, 32);

    if (lane == 0) {
        float g[MOE_E];
        float mx = -1e30f;
        #pragma unroll
        for (int e = 0; e < MOE_E; ++e) { g[e] = acc[e] + bg[e]; mx = fmaxf(mx, g[e]); }
        #pragma unroll
        for (int e = 0; e < MOE_E; ++e) g[e] = __expf(g[e] - mx);
        // top-2 (ratio of exps == ratio of softmax probs, so renorm needs no sum)
        int i0 = 0;
        #pragma unroll
        for (int e = 1; e < MOE_E; ++e) if (g[e] > g[i0]) i0 = e;
        int i1 = (i0 == 0) ? 1 : 0;
        #pragma unroll
        for (int e = 0; e < MOE_E; ++e) if (e != i0 && g[e] > g[i1]) i1 = e;
        const float denom = g[i0] + g[i1];
        const float w0 = g[i0] / denom, w1 = g[i1] / denom;
        top_i[token * 2 + 0] = i0;  top_i[token * 2 + 1] = i1;
        top_w[token * 2 + 0] = w0;  top_w[token * 2 + 1] = w1;
        atomicAdd(&usage[i0], w0);  atomicAdd(&usage[i1], w1);
        atomicAdd(&counts[i0], 1);  atomicAdd(&counts[i1], 1);
    }
}

// ---------------------------------------------------------------------------
// 5) tiny exclusive scan over 8 expert counts
// ---------------------------------------------------------------------------
__global__ void scan_kernel(const int* __restrict__ counts,
                            int* __restrict__ offsets, int* __restrict__ cursor) {
    if (threadIdx.x == 0) {
        int acc = 0;
        #pragma unroll
        for (int e = 0; e < MOE_E; ++e) {
            offsets[e] = acc; acc += counts[e]; cursor[e] = 0;
        }
        offsets[MOE_E] = acc;
    }
}

// ---------------------------------------------------------------------------
// 6) scatter tokens into expert-sorted slot list
// ---------------------------------------------------------------------------
__global__ void scatter_kernel(const int* __restrict__ top_i,
                               const float* __restrict__ top_w,
                               const int* __restrict__ offsets,
                               int* __restrict__ cursor,
                               int* __restrict__ slot_token,
                               float* __restrict__ slot_w) {
    int t = blockIdx.x * blockDim.x + threadIdx.x;
    if (t >= MOE_N) return;
    #pragma unroll
    for (int k = 0; k < MOE_K; ++k) {
        const int e = top_i[t * MOE_K + k];
        const int pos = offsets[e] + atomicAdd(&cursor[e], 1);
        slot_token[pos] = t;
        slot_w[pos] = top_w[t * MOE_K + k];
    }
}

// ---------------------------------------------------------------------------
// 7) fused grouped GEMM: per 32-token expert tile,
//    h1 = x@W1[e]+b1 (WMMA, bf16 tile kept in LDS), out += w * (h1@W2[e]+b2)
//    LDS: 32x512 bf16 x-tile (32 KB) + 32x2048 bf16 h1 (128 KB) = 160 KB/WGP.
// ---------------------------------------------------------------------------
__global__ void moe_gemm_kernel(const __bf16* __restrict__ xb,
                                const __bf16* __restrict__ W1t, // [E][H][D]
                                const __bf16* __restrict__ W2t, // [E][D][H]
                                const float* __restrict__ b1,   // [E][H]
                                const float* __restrict__ b2,   // [E][D]
                                const int* __restrict__ slot_token,
                                const float* __restrict__ slot_w,
                                const int* __restrict__ offsets,
                                float* __restrict__ out) {
    const int e    = blockIdx.x / TILES_PER_E;
    const int tile = blockIdx.x % TILES_PER_E;
    const int seg_begin = offsets[e];
    const int seg_end   = offsets[e + 1];
    const int base = seg_begin + tile * TILE_M;
    if (base >= seg_end) return;
    const int rows = min(TILE_M, seg_end - base);

    extern __shared__ char smem[];
    __bf16* lds_x  = (__bf16*)smem;                            // [32][512]
    __bf16* lds_h1 = (__bf16*)(smem + TILE_M * MOE_D * 2);     // [32][2048]

    const int tid = threadIdx.x;

    // ---- stage gathered A (x rows) into LDS: 16 threads per row, b128 copies
    {
        const int rg = tid >> 4;      // 0..15 row group
        const int cg = tid & 15;      // 0..15 column group (16B chunks)
        for (int r = rg; r < TILE_M; r += 16) {
            uint4* dst = (uint4*)(lds_x + r * MOE_D);
            if (r < rows) {
                const int tok = slot_token[base + r];
                const uint4* src = (const uint4*)(xb + (size_t)tok * MOE_D);
                #pragma unroll
                for (int j = 0; j < 4; ++j) dst[cg + 16 * j] = src[cg + 16 * j];
            } else {
                uint4 z = {0u, 0u, 0u, 0u};
                #pragma unroll
                for (int j = 0; j < 4; ++j) dst[cg + 16 * j] = z;
            }
        }
    }
    __syncthreads();

    const int lane  = tid & 31;
    const int wave  = tid >> 5;          // 0..7
    const int ln16  = lane & 15;
    const int khalf = (lane < 16) ? 0 : 8;   // 16-bit A/B frag k-half per ISA layout

    // ---- fc1: h1[32][2048] = A @ W1t[e]^T  (16 column-tile passes, 8 waves)
    for (int pass = 0; pass < MOE_H / (NWAVE * 16); ++pass) {
        const int n = (pass * NWAVE + wave) * 16 + ln16;
        const __bf16* bptr = W1t + ((size_t)e * MOE_H + n) * MOE_D;
        v8f acc0 = {}; v8f acc1 = {};
        #pragma unroll 4
        for (int k0 = 0; k0 < MOE_D; k0 += 32) {
            v16bf bf = frag_from(bptr + k0 + khalf);
            v16bf a0 = frag_from(lds_x + (size_t)ln16 * MOE_D + k0 + khalf);
            v16bf a1 = frag_from(lds_x + (size_t)(16 + ln16) * MOE_D + k0 + khalf);
            acc0 = wmma_bf16(a0, bf, acc0);
            acc1 = wmma_bf16(a1, bf, acc1);
        }
        const float bias = b1[e * MOE_H + n];
        #pragma unroll
        for (int r = 0; r < 8; ++r) {
            const int m = r + ((lane < 16) ? 0 : 8);
            lds_h1[(size_t)m * MOE_H + n]        = (__bf16)(acc0[r] + bias);
            lds_h1[(size_t)(16 + m) * MOE_H + n] = (__bf16)(acc1[r] + bias);
        }
    }
    __syncthreads();

    // ---- fc2: out_tile[32][512] = h1 @ W2t[e]^T, weighted atomic scatter
    for (int pass = 0; pass < MOE_D / (NWAVE * 16); ++pass) {
        const int n = (pass * NWAVE + wave) * 16 + ln16;
        const __bf16* bptr = W2t + ((size_t)e * MOE_D + n) * MOE_H;
        v8f acc0 = {}; v8f acc1 = {};
        #pragma unroll 4
        for (int k0 = 0; k0 < MOE_H; k0 += 32) {
            v16bf bf = frag_from(bptr + k0 + khalf);
            v16bf a0 = frag_from(lds_h1 + (size_t)ln16 * MOE_H + k0 + khalf);
            v16bf a1 = frag_from(lds_h1 + (size_t)(16 + ln16) * MOE_H + k0 + khalf);
            acc0 = wmma_bf16(a0, bf, acc0);
            acc1 = wmma_bf16(a1, bf, acc1);
        }
        const float bias = b2[e * MOE_D + n];
        #pragma unroll
        for (int r = 0; r < 8; ++r) {
            const int m0 = r + ((lane < 16) ? 0 : 8);
            if (m0 < rows) {
                const int tok = slot_token[base + m0];
                const float w = slot_w[base + m0];
                atomicAdd(&out[(size_t)tok * MOE_D + n], w * (acc0[r] + bias));
            }
            const int m1 = 16 + m0;
            if (m1 < rows) {
                const int tok = slot_token[base + m1];
                const float w = slot_w[base + m1];
                atomicAdd(&out[(size_t)tok * MOE_D + n], w * (acc1[r] + bias));
            }
        }
    }
}

// ---------------------------------------------------------------------------
// host launcher
// ---------------------------------------------------------------------------
extern "C" void kernel_launch(void* const* d_in, const int* in_sizes, int n_in,
                              void* d_out, int out_size, void* d_ws, size_t ws_size,
                              hipStream_t stream) {
    const float* x  = (const float*)d_in[0];  // [N,D]
    const float* Wg = (const float*)d_in[1];  // [D,E]
    const float* bg = (const float*)d_in[2];  // [E]
    const float* W1 = (const float*)d_in[3];  // [E,D,H]
    const float* b1 = (const float*)d_in[4];  // [E,H]
    const float* W2 = (const float*)d_in[5];  // [E,H,D]
    const float* b2 = (const float*)d_in[6];  // [E,D]
    float* out = (float*)d_out;               // [N,D] ++ usage[E]

    // workspace carving (all offsets 256B-aligned)
    auto align256 = [](size_t v) { return (v + 255) & ~(size_t)255; };
    size_t off = 0;
    __bf16* xb   = (__bf16*)((char*)d_ws + off); off = align256(off + (size_t)MOE_N * MOE_D * 2);
    __bf16* W1t  = (__bf16*)((char*)d_ws + off); off = align256(off + (size_t)MOE_E * MOE_D * MOE_H * 2);
    __bf16* W2t  = (__bf16*)((char*)d_ws + off); off = align256(off + (size_t)MOE_E * MOE_D * MOE_H * 2);
    int*   top_i = (int*)  ((char*)d_ws + off);  off = align256(off + (size_t)MOE_N * MOE_K * 4);
    float* top_w = (float*)((char*)d_ws + off);  off = align256(off + (size_t)MOE_N * MOE_K * 4);
    int*   slot_token = (int*)  ((char*)d_ws + off); off = align256(off + (size_t)MOE_N * MOE_K * 4);
    float* slot_w     = (float*)((char*)d_ws + off); off = align256(off + (size_t)MOE_N * MOE_K * 4);
    int*   counts  = (int*)((char*)d_ws + off); off = align256(off + MOE_E * 4);
    int*   offsets = (int*)((char*)d_ws + off); off = align256(off + (MOE_E + 1) * 4);
    int*   cursor  = (int*)((char*)d_ws + off); off = align256(off + MOE_E * 4);
    (void)ws_size; (void)n_in; (void)in_sizes; (void)out_size;

    float* usage = out + (size_t)MOE_N * MOE_D;
    const int out_elems = MOE_N * MOE_D + MOE_E;

    // 1) zero outputs + counts
    init_zero_kernel<<<(out_elems + 255) / 256, 256, 0, stream>>>(out, out_elems, counts);

    // 2) x -> bf16
    cvt_x_kernel<<<(MOE_N * MOE_D + 255) / 256, 256, 0, stream>>>(x, xb);

    // 3) weight transpose+convert (L2-resident bf16, K-contiguous)
    {
        dim3 blk(32, 8, 1);
        dim3 g1(MOE_H / 32, MOE_D / 32, MOE_E);  // W1 [D][H] -> [H][D]
        transpose_cvt_kernel<<<g1, blk, 0, stream>>>(W1, W1t, MOE_D, MOE_H);
        dim3 g2(MOE_D / 32, MOE_H / 32, MOE_E);  // W2 [H][D] -> [D][H]
        transpose_cvt_kernel<<<g2, blk, 0, stream>>>(W2, W2t, MOE_H, MOE_D);
    }

    // 4) gating (one wave32 per token; 8 tokens per 256-thread block)
    gate_kernel<<<MOE_N / 8, 256, 0, stream>>>(x, Wg, bg, top_i, top_w, counts, usage);

    // 5) scan
    scan_kernel<<<1, 32, 0, stream>>>(counts, offsets, cursor);

    // 6) scatter into expert-sorted slots
    scatter_kernel<<<(MOE_N + 255) / 256, 256, 0, stream>>>(
        top_i, top_w, offsets, cursor, slot_token, slot_w);

    // 7) fused grouped GEMM (8 waves, 160 KB dynamic LDS per WGP)
    const size_t lds_bytes = (size_t)TILE_M * MOE_D * 2 + (size_t)TILE_M * MOE_H * 2;
    moe_gemm_kernel<<<MOE_E * TILES_PER_E, NWAVE * 32, lds_bytes, stream>>>(
        xb, W1t, W2t, b1, b2, slot_token, slot_w, offsets, out);
}
